// GandACMLRegression_59373627900460
// MI455X (gfx1250) — compile-verified, matrix operand order checked by reading, not run on hardware
//
#include <hip/hip_runtime.h>

typedef unsigned short u16;
typedef unsigned int   u32;
typedef __attribute__((ext_vector_type(16))) __bf16 v16bf;
typedef __attribute__((ext_vector_type(8)))  float  v8f;

#define N_ROWS 32768
#define EMB    1024
#define HID    512

__device__ inline u16 f2bf(float f) {
  u32 u = __float_as_uint(f);
  u += 0x7FFFu + ((u >> 16) & 1u);     // round-to-nearest-even
  return (u16)(u >> 16);
}

struct Frag {
  union { v16bf v; uint4 q[2]; u16 u[16]; };
};

// bf16 row-major source: per-lane row pointer, K-tile k0, half = lane>>4.
// Lanes 0-15 take K = {k0..k0+7, k0+16..k0+23}; lanes 16-31 the other halves,
// matching the 16-bit A/B 16x32 VGPR striping in the CDNA5 ISA tables.
__device__ inline Frag load_frag_bf16(const u16* __restrict__ rowptr, int k0, int half) {
  Frag f;
  f.q[0] = *(const uint4*)(rowptr + k0 + half * 8);
  f.q[1] = *(const uint4*)(rowptr + k0 + 16 + half * 8);
  return f;
}

__device__ inline v8f vzero() {
  v8f v;
#pragma unroll
  for (int i = 0; i < 8; ++i) v[i] = 0.0f;
  return v;
}

__device__ inline v8f wmma_bf16(const Frag& a, const Frag& b, v8f c) {
  return __builtin_amdgcn_wmma_f32_16x16x32_bf16(
      /*neg_a=*/false, a.v, /*neg_b=*/false, b.v,
      /*c_mod=*/(short)0, c, /*reuse_a=*/false, /*reuse_b=*/false);
}

// ---------------- precompute kernels ----------------

// Vectorized f32 -> bf16: 4 elements/thread (float4 in, uint2 out). n4 = n/4.
__global__ void k_cvt4(const float* __restrict__ src, u16* __restrict__ dst, int n4) {
  int i = blockIdx.x * blockDim.x + threadIdx.x;
  if (i < n4) {
    float4 f = ((const float4*)src)[i];
    uint2 o;
    o.x = (u32)f2bf(f.x) | ((u32)f2bf(f.y) << 16);
    o.y = (u32)f2bf(f.z) | ((u32)f2bf(f.w) << 16);
    ((uint2*)dst)[i] = o;
  }
}

__global__ void k_bnprep(const float* __restrict__ b1, const float* __restrict__ gamma,
                         const float* __restrict__ beta, const float* __restrict__ rmean,
                         const float* __restrict__ rvar,
                         float* __restrict__ scale, float* __restrict__ shift) {
  int j = threadIdx.x;
  if (j < HID) {
    float sc = gamma[j] * rsqrtf(rvar[j] + 1e-5f);
    scale[j] = sc;
    shift[j] = (b1[j] - rmean[j]) * sc + beta[j];
  }
}

// ---------------- layer 1: h = leakyrelu(bn(x @ W1^T + b1)) ----------------

__device__ inline void epi_store(v8f acc, int j, const float* __restrict__ scale,
                                 const float* __restrict__ shift,
                                 u16* __restrict__ hout, int m0, int half) {
  float sc = scale[j], sh = shift[j];
#pragma unroll
  for (int i = 0; i < 8; ++i) {
    float v = acc[i] * sc + sh;
    v = fmaxf(v, 0.1f * v);                 // LeakyReLU(0.1)
    hout[(size_t)(m0 + i + half * 8) * HID + j] = f2bf(v);
  }
}

__global__ void __launch_bounds__(128) k_gemm1(
    const u16* __restrict__ xb,            // [3][N_ROWS][EMB] bf16
    const u16* __restrict__ W1b, const float* __restrict__ scale,
    const float* __restrict__ shift, u16* __restrict__ hbuf) {
  const int tid  = threadIdx.x;
  const int lane = tid & 31, wave = tid >> 5;
  const int unit = blockIdx.x * 4 + wave;     // 3 * 2048 units
  const int s    = unit >> 11;
  const int mt   = unit & 2047;
  const int m0   = mt * 16;
  const int half = lane >> 4, rn = lane & 15;
  const u16* xrow = xb + ((size_t)s * N_ROWS + m0 + rn) * EMB;
  u16* hout = hbuf + (size_t)s * N_ROWS * HID;

  for (int jt0 = 0; jt0 < 32; jt0 += 4) {
    v8f acc0 = vzero(), acc1 = vzero(), acc2 = vzero(), acc3 = vzero();
    const u16* w0 = W1b + (size_t)((jt0 + 0) * 16 + rn) * EMB;
    const u16* w1 = W1b + (size_t)((jt0 + 1) * 16 + rn) * EMB;
    const u16* w2 = W1b + (size_t)((jt0 + 2) * 16 + rn) * EMB;
    const u16* w3 = W1b + (size_t)((jt0 + 3) * 16 + rn) * EMB;
    // software pipeline: fragments for step kt+1 are in flight while the
    // WMMAs for step kt execute
    Frag a  = load_frag_bf16(xrow, 0, half);
    Frag b0 = load_frag_bf16(w0, 0, half);
    Frag b1f = load_frag_bf16(w1, 0, half);
    Frag b2f = load_frag_bf16(w2, 0, half);
    Frag b3f = load_frag_bf16(w3, 0, half);
#pragma unroll 4
    for (int kt = 0; kt < 31; ++kt) {
      const int k1 = (kt + 1) * 32;
      Frag a_n  = load_frag_bf16(xrow, k1, half);
      Frag b0_n = load_frag_bf16(w0, k1, half);
      Frag b1_n = load_frag_bf16(w1, k1, half);
      Frag b2_n = load_frag_bf16(w2, k1, half);
      Frag b3_n = load_frag_bf16(w3, k1, half);
      acc0 = wmma_bf16(a, b0, acc0);
      acc1 = wmma_bf16(a, b1f, acc1);
      acc2 = wmma_bf16(a, b2f, acc2);
      acc3 = wmma_bf16(a, b3f, acc3);
      a = a_n; b0 = b0_n; b1f = b1_n; b2f = b2_n; b3f = b3_n;
    }
    acc0 = wmma_bf16(a, b0, acc0);
    acc1 = wmma_bf16(a, b1f, acc1);
    acc2 = wmma_bf16(a, b2f, acc2);
    acc3 = wmma_bf16(a, b3f, acc3);

    epi_store(acc0, (jt0 + 0) * 16 + rn, scale, shift, hout, m0, half);
    epi_store(acc1, (jt0 + 1) * 16 + rn, scale, shift, hout, m0, half);
    epi_store(acc2, (jt0 + 2) * 16 + rn, scale, shift, hout, m0, half);
    epi_store(acc3, (jt0 + 3) * 16 + rn, scale, shift, hout, m0, half);
  }
}

// ------- layer 2 fused with distance reductions: e = h @ W2^T + b2 -------

__global__ void __launch_bounds__(128) k_gemm2(
    const u16* __restrict__ hbuf, const u16* __restrict__ W2b,
    const float* __restrict__ b2,
    float* __restrict__ d2p, float* __restrict__ d2n, float* __restrict__ rowsq) {
  const int tid  = threadIdx.x;
  const int lane = tid & 31, wave = tid >> 5;
  const int mt   = blockIdx.x * 4 + wave;     // 2048 m-tiles
  const int m0   = mt * 16;
  const int half = lane >> 4, rn = lane & 15;
  const u16* ra = hbuf + (size_t)(0 * N_ROWS + m0 + rn) * HID;
  const u16* rp = hbuf + (size_t)(1 * N_ROWS + m0 + rn) * HID;
  const u16* rg = hbuf + (size_t)(2 * N_ROWS + m0 + rn) * HID;

  float pp[8], nn[8], sq[8];
#pragma unroll
  for (int i = 0; i < 8; ++i) { pp[i] = 0.f; nn[i] = 0.f; sq[i] = 0.f; }

  for (int ot = 0; ot < 64; ++ot) {
    v8f aA = vzero(), aP = vzero(), aN = vzero();
    const u16* wrow = W2b + (size_t)(ot * 16 + rn) * HID;
    Frag b  = load_frag_bf16(wrow, 0, half);
    Frag fa = load_frag_bf16(ra, 0, half);
    Frag fp = load_frag_bf16(rp, 0, half);
    Frag fn = load_frag_bf16(rg, 0, half);
#pragma unroll 5
    for (int jt = 0; jt < 15; ++jt) {
      const int j1 = (jt + 1) * 32;
      Frag b_n  = load_frag_bf16(wrow, j1, half);
      Frag fa_n = load_frag_bf16(ra, j1, half);
      Frag fp_n = load_frag_bf16(rp, j1, half);
      Frag fn_n = load_frag_bf16(rg, j1, half);
      aA = wmma_bf16(fa, b, aA);
      aP = wmma_bf16(fp, b, aP);
      aN = wmma_bf16(fn, b, aN);
      b = b_n; fa = fa_n; fp = fp_n; fn = fn_n;
    }
    aA = wmma_bf16(fa, b, aA);
    aP = wmma_bf16(fp, b, aP);
    aN = wmma_bf16(fn, b, aN);

    float bias = b2[ot * 16 + rn];
#pragma unroll
    for (int i = 0; i < 8; ++i) {
      float ea = aA[i] + bias, ep = aP[i] + bias, en = aN[i] + bias;
      float dp = ea - ep, dn = ea - en;
      pp[i] += dp * dp;
      nn[i] += dn * dn;
      sq[i] += ea * ea + ep * ep + en * en;
    }
  }
  // C/D layout: VGPR i holds row i (lanes 0-15) / row i+8 (lanes 16-31), col = lane&15.
  // Reduce across the 16 lanes of each half (masks <= 8 never cross halves).
#pragma unroll
  for (int m = 1; m <= 8; m <<= 1) {
#pragma unroll
    for (int i = 0; i < 8; ++i) {
      pp[i] += __shfl_xor(pp[i], m, 32);
      nn[i] += __shfl_xor(nn[i], m, 32);
      sq[i] += __shfl_xor(sq[i], m, 32);
    }
  }
  if (rn == 0) {                                // lanes 0 and 16
#pragma unroll
    for (int i = 0; i < 8; ++i) {
      int row = m0 + i + half * 8;
      d2p[row] = pp[i];
      d2n[row] = nn[i];
      rowsq[row] = sq[i];
    }
  }
}

// ---------------- margins, quantile, final loss ----------------

__global__ void k_margins(const float* __restrict__ d2p, const float* __restrict__ d2n,
                          float* __restrict__ margins) {
  int i = blockIdx.x * blockDim.x + threadIdx.x;
  if (i < N_ROWS) {
    float dp = sqrtf(d2p[i]), dn = sqrtf(d2n[i]);
    margins[i] = (dn < dp) ? (dp - dn) : 0.0f;
  }
}

// Single-block MSB-first binary radix select; margins >= 0 so uint bits are
// order-isomorphic to float values. Finds ranks 8191 & 8192 (0-indexed),
// then Q1 with linear interpolation: pos = 0.25*(N-1) = 8191.75.
__global__ void k_quantile(const float* __restrict__ margins, float* __restrict__ out_margin) {
  __shared__ u32 s_cnt, s_prefix, s_k;
  const int tid = threadIdx.x;
  float vals[2];
  for (int r = 0; r < 2; ++r) {
    if (tid == 0) { s_prefix = 0u; s_k = 8191u + (u32)r; }
    __syncthreads();
    for (int bit = 31; bit >= 0; --bit) {
      if (tid == 0) s_cnt = 0u;
      __syncthreads();
      u32 mask = (bit == 31) ? 0u : (~0u << (bit + 1));
      u32 pref = s_prefix;
      u32 local = 0;
      for (int i = tid; i < N_ROWS; i += 256) {
        u32 u = __float_as_uint(margins[i]);
        if ((u & mask) == pref && ((u >> bit) & 1u) == 0u) local++;
      }
      atomicAdd(&s_cnt, local);
      __syncthreads();
      if (tid == 0) {
        if (s_k < s_cnt) { /* bit stays 0 */ }
        else { s_prefix |= (1u << bit); s_k -= s_cnt; }
      }
      __syncthreads();
    }
    vals[r] = __uint_as_float(s_prefix);
    __syncthreads();
  }
  if (tid == 0) *out_margin = vals[0] + 0.75f * (vals[1] - vals[0]);
}

__global__ void k_loss(const float* __restrict__ d2p, const float* __restrict__ d2n,
                       const float* __restrict__ rowsq, const float* __restrict__ marginp,
                       float* __restrict__ out) {
  __shared__ float r1[256], r2[256];
  const int tid = threadIdx.x;
  const float margin = *marginp;
  float tsum = 0.f, ssum = 0.f;
  for (int i = tid; i < N_ROWS; i += 256) {
    float t = d2n[i] - d2p[i] + margin;       // dist_neg^2 - dist_pos^2 + margin
    tsum += (t > 0.f) ? t : 0.f;
    ssum += rowsq[i];
  }
  r1[tid] = tsum; r2[tid] = ssum;
  __syncthreads();
  for (int s = 128; s > 0; s >>= 1) {
    if (tid < s) { r1[tid] += r1[tid + s]; r2[tid] += r2[tid + s]; }
    __syncthreads();
  }
  if (tid == 0)
    out[0] = r1[0] / (float)N_ROWS + 0.001f * r2[0] / (float)N_ROWS;
}

// ---------------- launcher ----------------

extern "C" void kernel_launch(void* const* d_in, const int* in_sizes, int n_in,
                              void* d_out, int out_size, void* d_ws, size_t ws_size,
                              hipStream_t stream) {
  (void)in_sizes; (void)n_in; (void)out_size; (void)ws_size;
  const float* xA    = (const float*)d_in[0];
  const float* xP    = (const float*)d_in[1];
  const float* xN    = (const float*)d_in[2];
  const float* W1    = (const float*)d_in[3];
  const float* b1    = (const float*)d_in[4];
  const float* gamma = (const float*)d_in[5];
  const float* beta  = (const float*)d_in[6];
  const float* rmean = (const float*)d_in[7];
  const float* rvar  = (const float*)d_in[8];
  const float* W2    = (const float*)d_in[9];
  const float* b2    = (const float*)d_in[10];

  char* ws = (char*)d_ws;
  u16*   W1b     = (u16*)ws;                          // 1 MB
  u16*   W2b     = (u16*)(ws + (1 << 20));            // 1 MB
  float* scale   = (float*)(ws + (2 << 20));          // 2 KB
  float* shift   = scale + HID;                       // 2 KB
  float* d2p     = (float*)(ws + (2 << 20) + 8192);   // 128 KB
  float* d2n     = d2p + N_ROWS;                      // 128 KB
  float* rowsq   = d2n + N_ROWS;                      // 128 KB
  float* margins = rowsq + N_ROWS;                    // 128 KB
  float* marginv = margins + N_ROWS;                  // 4 B
  u16*   hbuf    = (u16*)(ws + ((size_t)4 << 20));    // 3*32768*512*2 = 96 MB
  u16*   xb      = (u16*)(ws + ((size_t)100 << 20));  // 3*32768*1024*2 = 192 MB

  const int xn4 = N_ROWS * EMB / 4;                   // 8,388,608 per stream
  const int wn4 = HID * EMB / 4;                      // 131,072
  k_cvt4<<<xn4 / 256, 256, 0, stream>>>(xA, xb + (size_t)0 * N_ROWS * EMB, xn4);
  k_cvt4<<<xn4 / 256, 256, 0, stream>>>(xP, xb + (size_t)1 * N_ROWS * EMB, xn4);
  k_cvt4<<<xn4 / 256, 256, 0, stream>>>(xN, xb + (size_t)2 * N_ROWS * EMB, xn4);
  k_cvt4<<<wn4 / 256, 256, 0, stream>>>(W1, W1b, wn4);
  k_cvt4<<<wn4 / 256, 256, 0, stream>>>(W2, W2b, wn4);
  k_bnprep<<<1, 512, 0, stream>>>(b1, gamma, beta, rmean, rvar, scale, shift);
  k_gemm1<<<1536, 128, 0, stream>>>(xb, W1b, scale, shift, hbuf);     // 6144 waves
  k_gemm2<<<512, 128, 0, stream>>>(hbuf, W2b, b2, d2p, d2n, rowsq);   // 2048 waves
  k_margins<<<128, 256, 0, stream>>>(d2p, d2n, margins);
  k_quantile<<<1, 256, 0, stream>>>(margins, marginv);
  k_loss<<<1, 256, 0, stream>>>(d2p, d2n, rowsq, marginv, (float*)d_out);
}
